// MaskedRNN_15118284881968
// MI455X (gfx1250) — matrix operand order repaired and checked by reading, not compiled, not run on hardware
//
#include <hip/hip_runtime.h>
#include <hip/hip_bf16.h>

typedef __attribute__((ext_vector_type(16))) __bf16 v16bf;
typedef __attribute__((ext_vector_type(8)))  __bf16 v8bf;
typedef __attribute__((ext_vector_type(8)))  float  v8f;

#define NT   512      // timesteps
#define NB   64       // batch
#define ND   512      // input dim
#define NH   512      // hidden dim
#define NG   2048     // 4*H gate dim
#define NBLK 32       // persistent blocks (each owns 16 hidden units)
#define XKT  (ND / 32) // k-steps in phase-1 GEMM

__device__ __forceinline__ float sigf(float x) { return 1.0f / (1.0f + expf(-x)); }

// Load A fragment (16x32 bf16) from row-major [.,K] source.
// lane l: M = l%16, klo = (l<16?0:8): elems 0..7 = K[klo..klo+8), 8..15 = K[klo+16..+24)
__device__ __forceinline__ v16bf load_a(const __bf16* __restrict__ row, int kbase, int hi) {
    int klo = kbase + hi * 8;
    v8bf lo = *(const v8bf*)(row + klo);
    v8bf hh = *(const v8bf*)(row + klo + 16);
    return __builtin_shufflevector(lo, hh, 0, 1, 2, 3, 4, 5, 6, 7, 8, 9, 10, 11, 12, 13, 14, 15);
}

__device__ __forceinline__ v16bf cat8(v8bf lo, v8bf hh) {
    return __builtin_shufflevector(lo, hh, 0, 1, 2, 3, 4, 5, 6, 7, 8, 9, 10, 11, 12, 13, 14, 15);
}

// ---------- conversion / packing kernels ----------
__global__ __launch_bounds__(256) void f32_to_bf16_kern(const float* __restrict__ src,
                                                        __bf16* __restrict__ dst, int n) {
    int i = blockIdx.x * 256 + threadIdx.x;
    if (i < n) dst[i] = (__bf16)src[i];
}

// src [K][N] f32 -> dst [N][K] bf16 (transpose so WMMA B loads are contiguous per lane)
__global__ __launch_bounds__(256) void transpose_bf16_kern(const float* __restrict__ src,
                                                           __bf16* __restrict__ dst, int K, int N) {
    int i = blockIdx.x * 256 + threadIdx.x;
    if (i < K * N) {
        int k = i / N, n = i % N;
        dst[(size_t)n * K + k] = (__bf16)src[i];
    }
}

// h_buf[0] = bf16(h0 * (1 - mask[0]))  (pre-masked state entering step 0)
__global__ __launch_bounds__(256) void init_h_kern(const float* __restrict__ h0,
                                                   const float* __restrict__ mask,
                                                   __bf16* __restrict__ hb) {
    int i = blockIdx.x * 256 + threadIdx.x;
    if (i < NB * NH) {
        int b = i >> 9;  // /NH
        hb[i] = (__bf16)(h0[i] * (1.0f - mask[b]));
    }
}

// ---------- phase 1: x_proj = x @ Wx + b  (bf16 WMMA, fp32 accum) ----------
// B tile staged in LDS via async-to-LDS loads (double buffered), shared by all 8 waves.
// Xb [32768][512] bf16, Wxt [2048][512] bf16 (transposed), XP [32768][2048] f32
__global__ __launch_bounds__(256) void xproj_gemm_kern(const __bf16* __restrict__ Xb,
                                                       const __bf16* __restrict__ Wxt,
                                                       const float* __restrict__ bias,
                                                       float* __restrict__ XP) {
    // [2 bufs][64 cols][40 elems] bf16: stride 40 (80B = 20 banks) -> conflict-free frag reads
    __shared__ __attribute__((aligned(32))) __bf16 bs[2][64][40];

    const int tid  = threadIdx.x;
    const int lane = tid & 31;
    const int wave = tid >> 5;
    const int l15  = lane & 15;
    const int hi   = lane >> 4;
    const int m0   = blockIdx.x * 128 + wave * 16;  // 16 rows per wave
    const int n0   = blockIdx.y * 64;               // 64 cols per block

    // Staging: thread -> (col = tid/4, 8-elem segment = tid%4), one b128 async load per k-step
    const int scol = tid >> 2;
    const int sseg = (tid & 3) * 8;
    const __bf16* srcBase = Wxt + (size_t)(n0 + scol) * ND + sseg;

    auto stage = [&](int kt) {
        unsigned ldsOff = (unsigned)(size_t)(const void*)&bs[kt & 1][scol][sseg];
        unsigned long long ga = (unsigned long long)(size_t)(srcBase + kt * 32);
        asm volatile("global_load_async_to_lds_b128 %0, %1, off"
                     :: "v"(ldsOff), "v"(ga) : "memory");
    };

    v8f acc[4] = {};
    const __bf16* arow = Xb + (size_t)(m0 + l15) * ND;

    stage(0);
    for (int kt = 0; kt < XKT; ++kt) {
        if (kt + 1 < XKT) {
            stage(kt + 1);                                    // fill other buffer
            asm volatile("s_wait_asynccnt 0x1" ::: "memory"); // kt's tile landed in LDS
        } else {
            asm volatile("s_wait_asynccnt 0x0" ::: "memory");
        }
        __syncthreads();                                      // tile visible to all waves

        v16bf a = load_a(arow, kt * 32, hi);
        const int p = kt & 1;
        #pragma unroll
        for (int f = 0; f < 4; ++f) {
            v8bf blo = *(const v8bf*)&bs[p][f * 16 + l15][hi * 8];
            v8bf bhi = *(const v8bf*)&bs[p][f * 16 + l15][hi * 8 + 16];
            v16bf b = cat8(blo, bhi);
            acc[f] = __builtin_amdgcn_wmma_f32_16x16x32_bf16(false, a, false, b, (short)0,
                                                             acc[f], false, false);
        }
        __syncthreads();  // done reading buf[p] before it is refilled at kt+2
    }

    #pragma unroll
    for (int f = 0; f < 4; ++f) {
        int col = n0 + f * 16 + l15;
        float bv = bias[col];
        #pragma unroll
        for (int r = 0; r < 8; ++r) {
            int row = m0 + r + hi * 8;
            XP[(size_t)row * NG + col] = acc[f][r] + bv;
        }
    }
}

// ---------- phase 2: persistent masked-LSTM scan ----------
// Block bx owns hidden cols [bx*16, bx*16+16). Wh slice lives in LDS all 512 steps,
// c slice lives in registers. One grid barrier per step; h double-buffered in global (L2).
__global__ __launch_bounds__(256) void lstm_scan_kern(const float* __restrict__ XP,
                                                      const float* __restrict__ mask,
                                                      const __bf16* __restrict__ Wht,
                                                      const float* __restrict__ c0,
                                                      __bf16* __restrict__ hbuf,  // [2][64][512]
                                                      float* __restrict__ out,    // outputs|hfin|cfin
                                                      unsigned* __restrict__ bar) {
    extern __shared__ char smem[];
    __bf16* whs  = (__bf16*)smem;                        // [64][512] bf16 = 64 KB
    float*  gbuf = (float*)(smem + 64 * NH * 2);         // [4][64][16] f32 = 16 KB

    const int tid   = threadIdx.x;
    const int lane  = tid & 31;
    const int wave  = tid >> 5;
    const int l15   = lane & 15;
    const int hi    = lane >> 4;
    const int hbase = blockIdx.x * 16;

    // Preload this block's Wh slice (4 gates x 16 cols x 512 K) into LDS, transposed layout.
    for (int i = tid; i < 64 * (NH / 8); i += 256) {
        int lc = i >> 6;              // local col 0..63  (gate = lc>>4, col = lc&15)
        int ko = (i & 63) * 8;
        int gcol = (lc >> 4) * NH + hbase + (lc & 15);
        *(v8bf*)&whs[lc * NH + ko] = *(const v8bf*)&Wht[(size_t)gcol * NH + ko];
    }

    // c-state: each thread owns 4 cells (batch row cb, cols cc..cc+3)
    const int cb = tid >> 2;
    const int cc = (tid & 3) * 4;
    float c[4], hreg[4] = {0.f, 0.f, 0.f, 0.f};
    #pragma unroll
    for (int j = 0; j < 4; ++j) c[j] = c0[cb * NH + hbase + cc + j];
    __syncthreads();

    const int mt   = wave & 3;           // batch tile (rows mt*16..)
    const int g0   = (wave >> 2) * 2;    // this wave computes gates g0, g0+1
    const int arow = mt * 16 + l15;

    for (int t = 0; t < NT; ++t) {
        // Prefetch next step's x_proj slice (64 rows x 4 gate segments of 64B = 256 lines):
        // hides HBM latency under this step's GEMM + grid barrier.
        if (t + 1 < NT) {
            const float* pf = XP + ((size_t)(t + 1) * NB + (tid & 63)) * NG
                              + (tid >> 6) * NH + hbase;
            __builtin_prefetch(pf, 0, 1);
        }

        const __bf16* hb = hbuf + (size_t)(t & 1) * (NB * NH);
        v8f acc0 = {}, acc1 = {};
        #pragma unroll 4
        for (int kt = 0; kt < NH / 32; ++kt) {
            v16bf a = load_a(hb + (size_t)arow * NH, kt * 32, hi);
            int koff = kt * 32 + hi * 8;
            v16bf b0 = cat8(*(const v8bf*)&whs[(size_t)((g0 + 0) * 16 + l15) * NH + koff],
                            *(const v8bf*)&whs[(size_t)((g0 + 0) * 16 + l15) * NH + koff + 16]);
            v16bf b1 = cat8(*(const v8bf*)&whs[(size_t)((g0 + 1) * 16 + l15) * NH + koff],
                            *(const v8bf*)&whs[(size_t)((g0 + 1) * 16 + l15) * NH + koff + 16]);
            acc0 = __builtin_amdgcn_wmma_f32_16x16x32_bf16(false, a, false, b0, (short)0,
                                                           acc0, false, false);
            acc1 = __builtin_amdgcn_wmma_f32_16x16x32_bf16(false, a, false, b1, (short)0,
                                                           acc1, false, false);
        }
        // scatter gate fragments to LDS
        #pragma unroll
        for (int r = 0; r < 8; ++r) {
            int row = mt * 16 + r + hi * 8;
            gbuf[((g0 + 0) * 64 + row) * 16 + l15] = acc0[r];
            gbuf[((g0 + 1) * 64 + row) * 16 + l15] = acc1[r];
        }
        __syncthreads();

        // cell update (gates = h@Wh (LDS) + x_proj (prefetched, includes bias))
        const float* xp = XP + ((size_t)t * NB + cb) * NG;
        float keep   = 1.0f - mask[t * NB + cb];
        float keep_n = (t + 1 < NT) ? (1.0f - mask[(t + 1) * NB + cb]) : 1.0f;
        #pragma unroll
        for (int j = 0; j < 4; ++j) {
            int col = cc + j;
            float gi = gbuf[(0 * 64 + cb) * 16 + col] + xp[0 * NH + hbase + col];
            float gf = gbuf[(1 * 64 + cb) * 16 + col] + xp[1 * NH + hbase + col];
            float gg = gbuf[(2 * 64 + cb) * 16 + col] + xp[2 * NH + hbase + col];
            float go = gbuf[(3 * 64 + cb) * 16 + col] + xp[3 * NH + hbase + col];
            float cm = c[j] * keep;
            float cn = sigf(gf) * cm + sigf(gi) * tanhf(gg);
            float h  = sigf(go) * tanhf(cn);
            c[j]    = cn;
            hreg[j] = h;
            out[((size_t)t * NB + cb) * NH + hbase + col] = h;
        }
        // publish pre-masked bf16 h for step t+1 (double buffer)
        __bf16* hn = hbuf + (size_t)((t + 1) & 1) * (NB * NH) + (size_t)cb * NH + hbase + cc;
        #pragma unroll
        for (int j = 0; j < 4; ++j) hn[j] = (__bf16)(hreg[j] * keep_n);

        // device-wide barrier (monotonic counter; reset by memset each launch)
        __threadfence();
        __syncthreads();
        if (tid == 0) {
            __hip_atomic_fetch_add(bar, 1u, __ATOMIC_RELEASE, __HIP_MEMORY_SCOPE_AGENT);
            unsigned target = (unsigned)NBLK * (unsigned)(t + 1);
            while (__hip_atomic_load(bar, __ATOMIC_ACQUIRE, __HIP_MEMORY_SCOPE_AGENT) < target)
                __builtin_amdgcn_s_sleep(1);
        }
        __syncthreads();
    }

    // finals
    float* hfin = out + (size_t)NT * NB * NH;
    float* cfin = hfin + NB * NH;
    #pragma unroll
    for (int j = 0; j < 4; ++j) {
        hfin[cb * NH + hbase + cc + j] = hreg[j];
        cfin[cb * NH + hbase + cc + j] = c[j];
    }
}

extern "C" void kernel_launch(void* const* d_in, const int* in_sizes, int n_in,
                              void* d_out, int out_size, void* d_ws, size_t ws_size,
                              hipStream_t stream) {
    (void)in_sizes; (void)n_in; (void)out_size; (void)ws_size;
    const float* x    = (const float*)d_in[0];  // [T,B,D]
    const float* h0   = (const float*)d_in[1];  // [B,H]
    const float* c0   = (const float*)d_in[2];  // [B,H]
    const float* mask = (const float*)d_in[3];  // [T,B]
    const float* Wx   = (const float*)d_in[4];  // [D,4H]
    const float* Wh   = (const float*)d_in[5];  // [H,4H]
    const float* bias = (const float*)d_in[6];  // [4H]

    // workspace layout
    char*   w    = (char*)d_ws;
    float*  XP   = (float*)w;                       // 512*64*2048*4 = 256 MB
    size_t  off  = (size_t)NT * NB * NG * 4;
    __bf16* Wxt  = (__bf16*)(w + off); off += (size_t)NG * ND * 2;      // 2 MB
    __bf16* Wht  = (__bf16*)(w + off); off += (size_t)NG * NH * 2;      // 2 MB
    __bf16* Xb   = (__bf16*)(w + off); off += (size_t)NT * NB * ND * 2; // 32 MB
    __bf16* hbuf = (__bf16*)(w + off); off += (size_t)2 * NB * NH * 2;  // 128 KB
    unsigned* bar = (unsigned*)(w + off);

    hipMemsetAsync(bar, 0, 256, stream);

    int nx = NT * NB * ND;
    f32_to_bf16_kern<<<(nx + 255) / 256, 256, 0, stream>>>(x, Xb, nx);
    transpose_bf16_kern<<<(ND * NG + 255) / 256, 256, 0, stream>>>(Wx, Wxt, ND, NG);
    transpose_bf16_kern<<<(NH * NG + 255) / 256, 256, 0, stream>>>(Wh, Wht, NH, NG);
    init_h_kern<<<(NB * NH + 255) / 256, 256, 0, stream>>>(h0, mask, hbuf);

    // phase 1: M tiles 32768/128 = 256, N tiles 2048/64 = 32
    xproj_gemm_kern<<<dim3(256, 32), 256, 0, stream>>>(Xb, Wxt, bias, XP);

    // phase 2: persistent scan, 80 KB dynamic LDS (64 KB Wh slice + 16 KB gate exchange)
    size_t lds = (size_t)64 * NH * 2 + (size_t)4 * 64 * 16 * 4;
    lstm_scan_kern<<<NBLK, 256, lds, stream>>>(XP, mask, Wht, c0, hbuf, (float*)d_out, bar);
}
